// EdgeSAGERegressor_4243427688733
// MI455X (gfx1250) — compile-verified, matrix operand order checked by reading, not run on hardware
//
#include <hip/hip_runtime.h>
#include <hip/hip_bf16.h>

#define NN 50000
#define NE 800000
#define NG 50
#define BN_EPS 1e-5f

typedef __attribute__((ext_vector_type(16))) _Float16 v16h;
typedef __attribute__((ext_vector_type(8)))  _Float16 v8h;
typedef __attribute__((ext_vector_type(8)))  float    v8f;
typedef __attribute__((ext_vector_type(4)))  float    v4f;

__device__ __forceinline__ v16h cat8(v8h lo, v8h hi) {
  return __builtin_shufflevector(lo, hi, 0,1,2,3,4,5,6,7,8,9,10,11,12,13,14,15);
}

// -------- zero fill --------
__global__ void zero_kernel(float* __restrict__ p, int n) {
  int i = blockIdx.x * blockDim.x + threadIdx.x;
  int stride = gridDim.x * blockDim.x;
  for (; i < n; i += stride) p[i] = 0.0f;
}

// -------- convert f32 node features -> f16 --------
__global__ __launch_bounds__(256) void cvt_kernel(const float* __restrict__ src,
                                                  _Float16* __restrict__ dst, int n) {
  int i = blockIdx.x * blockDim.x + threadIdx.x;
  int stride = gridDim.x * blockDim.x;
  for (; i < n; i += stride) dst[i] = (_Float16)src[i];
}

// -------- weights: transpose + f16 --------
__global__ void prep_weights_kernel(const float* __restrict__ Wn1, const float* __restrict__ Wn2,
                                    const float* __restrict__ Wr1, const float* __restrict__ Wr2,
                                    _Float16* __restrict__ Wn1T, _Float16* __restrict__ Wn2T,
                                    _Float16* __restrict__ Wr1T, _Float16* __restrict__ Wr2T) {
  int i = blockIdx.x * blockDim.x + threadIdx.x;
  if (i < 96 * 64) {
    int k = i >> 6, n = i & 63;
    Wn1T[n * 96 + k] = (_Float16)Wn1[i];
    Wn2T[n * 96 + k] = (_Float16)Wn2[i];
    if (k < 64) {
      Wr1T[n * 64 + k] = (_Float16)Wr1[i];
      Wr2T[n * 64 + k] = (_Float16)Wr2[i];
    }
  }
}

// -------- in-degree counts (for scatter-mean) --------
__global__ void deg_kernel(const int* __restrict__ dstIdx, float* __restrict__ deg) {
  int e = blockIdx.x * blockDim.x + threadIdx.x;
  if (e < NE) unsafeAtomicAdd(&deg[dstIdx[e]], 1.0f);
}

// -------- edge message GEMM + ReLU + fused scatter-add (core WMMA kernel) --------
// m = relu(concat(feat[src], eattr) @ Wn + bias) ; sOut[dst] += m
__global__ __launch_bounds__(256)
void edge_msg_wmma(const _Float16* __restrict__ feat,   // [NN,64] f16
                   const float* __restrict__ eattr,     // [NE,32] f32
                   const int* __restrict__ srcIdx, const int* __restrict__ dstIdx,
                   const _Float16* __restrict__ WT,     // [64,96] = Wn^T f16
                   const float* __restrict__ bias,      // [64]
                   float* __restrict__ sOut)            // [NN,64] f32 accum
{
  __shared__ _Float16 sW[64 * 96];
  for (int i = threadIdx.x; i < 64 * 96 * 2 / 16; i += blockDim.x)
    ((int4*)sW)[i] = ((const int4*)WT)[i];
  __syncthreads();

  const int lane = threadIdx.x & 31;
  const int wv   = threadIdx.x >> 5;
  const int l16  = lane & 15;
  const int hf   = lane >> 4;      // which half of the wave
  const int sel  = hf * 8;         // K sub-select per ISA 16-bit A/B layout

  float bcol[4];
#pragma unroll
  for (int nt = 0; nt < 4; ++nt) bcol[nt] = bias[nt * 16 + l16];

  const int nTiles  = NE / 16;
  const int wvGlob  = blockIdx.x * (blockDim.x >> 5) + wv;
  const int nWaves  = gridDim.x * (blockDim.x >> 5);

  for (int t = wvGlob; t < nTiles; t += nWaves) {
    const int eBase = t * 16;
    const int e   = eBase + l16;
    const int src = srcIdx[e];
    __builtin_prefetch(eattr + (size_t)(e + 16 * nWaves) * 32, 0, 1);

    v8f c[4] = {};
#pragma unroll
    for (int kt = 0; kt < 3; ++kt) {
      v16h a;
      if (kt < 2) {              // node features, already f16
        const _Float16* p = feat + src * 64 + kt * 32 + sel;
        a = cat8(*(const v8h*)p, *(const v8h*)(p + 16));
      } else {                   // edge_attr f32 -> f16 inline
        const float* p = eattr + (size_t)e * 32 + sel;
        v4f f0 = *(const v4f*)p;
        v4f f1 = *(const v4f*)(p + 4);
        v4f f2 = *(const v4f*)(p + 16);
        v4f f3 = *(const v4f*)(p + 20);
        v16h aa;
#pragma unroll
        for (int i = 0; i < 4; ++i) {
          aa[i]      = (_Float16)f0[i];
          aa[4 + i]  = (_Float16)f1[i];
          aa[8 + i]  = (_Float16)f2[i];
          aa[12 + i] = (_Float16)f3[i];
        }
        a = aa;
      }
#pragma unroll
      for (int nt = 0; nt < 4; ++nt) {
        const _Float16* q = sW + (nt * 16 + l16) * 96 + kt * 32 + sel;
        v16h b = cat8(*(const v8h*)q, *(const v8h*)(q + 16));
        c[nt] = __builtin_amdgcn_wmma_f32_16x16x32_f16(false, a, false, b,
                                                       (short)0, c[nt], false, false);
      }
    }

    int dstv[8];
#pragma unroll
    for (int r = 0; r < 8; ++r) dstv[r] = dstIdx[eBase + hf * 8 + r];
#pragma unroll
    for (int nt = 0; nt < 4; ++nt) {
      const int n = nt * 16 + l16;
#pragma unroll
      for (int r = 0; r < 8; ++r) {
        float v = fmaxf(c[nt][r] + bcol[nt], 0.0f);
        unsafeAtomicAdd(&sOut[dstv[r] * 64 + n], v);
      }
    }
  }
}

// -------- node update: h = feat @ Wr + s/deg  (WMMA, K=64) --------
__global__ __launch_bounds__(256)
void node_update_wmma(const _Float16* __restrict__ feat,  // [NN,64] f16
                      const _Float16* __restrict__ WT,    // [64,64] = Wr^T f16
                      const float* __restrict__ sIn,      // [NN,64]
                      const float* __restrict__ deg,      // [NN]
                      float* __restrict__ hOut)           // [NN,64]
{
  __shared__ _Float16 sW[64 * 64];
  for (int i = threadIdx.x; i < 64 * 64 * 2 / 16; i += blockDim.x)
    ((int4*)sW)[i] = ((const int4*)WT)[i];
  __syncthreads();

  const int lane = threadIdx.x & 31;
  const int wv   = threadIdx.x >> 5;
  const int l16  = lane & 15;
  const int hf   = lane >> 4;
  const int sel  = hf * 8;

  const int nTiles = NN / 16;
  const int wvGlob = blockIdx.x * (blockDim.x >> 5) + wv;
  const int nWaves = gridDim.x * (blockDim.x >> 5);

  for (int t = wvGlob; t < nTiles; t += nWaves) {
    const int nBase = t * 16;
    const int row   = nBase + l16;

    v8f c[4] = {};
#pragma unroll
    for (int kt = 0; kt < 2; ++kt) {
      const _Float16* p = feat + row * 64 + kt * 32 + sel;
      v16h a = cat8(*(const v8h*)p, *(const v8h*)(p + 16));
#pragma unroll
      for (int nt = 0; nt < 4; ++nt) {
        const _Float16* q = sW + (nt * 16 + l16) * 64 + kt * 32 + sel;
        v16h b = cat8(*(const v8h*)q, *(const v8h*)(q + 16));
        c[nt] = __builtin_amdgcn_wmma_f32_16x16x32_f16(false, a, false, b,
                                                       (short)0, c[nt], false, false);
      }
    }

    int rows[8];
    float invd[8];
#pragma unroll
    for (int r = 0; r < 8; ++r) {
      rows[r] = nBase + hf * 8 + r;
      invd[r] = 1.0f / fmaxf(deg[rows[r]], 1.0f);
    }
#pragma unroll
    for (int nt = 0; nt < 4; ++nt) {
      const int n = nt * 16 + l16;
#pragma unroll
      for (int r = 0; r < 8; ++r)
        hOut[rows[r] * 64 + n] = c[nt][r] + sIn[rows[r] * 64 + n] * invd[r];
    }
  }
}

// -------- BN stats (sum, sumsq per feature) --------
__global__ __launch_bounds__(256)
void bn_stats_kernel(const float* __restrict__ h, float* __restrict__ stats) {
  __shared__ float sh[512];
  const int tid = threadIdx.x;
  float s = 0.f, s2 = 0.f;
  const int total = gridDim.x * blockDim.x;   // multiple of 64 -> feature fixed per thread
  for (int i = blockIdx.x * blockDim.x + tid; i < NN * 64; i += total) {
    float v = h[i];
    s += v; s2 += v * v;
  }
  sh[tid] = s; sh[256 + tid] = s2;
  __syncthreads();
  if (tid < 64) {
    float a = sh[tid] + sh[tid + 64] + sh[tid + 128] + sh[tid + 192];
    float b = sh[256 + tid] + sh[256 + tid + 64] + sh[256 + tid + 128] + sh[256 + tid + 192];
    unsafeAtomicAdd(&stats[tid], a);
    unsafeAtomicAdd(&stats[64 + tid], b);
  }
}

__global__ void bn_final_kernel(const float* __restrict__ stats, const float* __restrict__ g,
                                const float* __restrict__ be, float* __restrict__ fin) {
  int f = threadIdx.x;
  if (f < 64) {
    float mean = stats[f] * (1.0f / (float)NN);
    float var  = stats[64 + f] * (1.0f / (float)NN) - mean * mean;
    float sc   = g[f] * rsqrtf(var + BN_EPS);
    fin[f]      = sc;
    fin[64 + f] = be[f] - mean * sc;
  }
}

// -------- apply BN + ReLU; also emit f16 copy for next layer's WMMA A --------
__global__ __launch_bounds__(256)
void bn_apply_kernel(float* __restrict__ h, _Float16* __restrict__ hh,
                     const float* __restrict__ fin) {
  int i = blockIdx.x * blockDim.x + threadIdx.x;
  const int stride = gridDim.x * blockDim.x;
  for (; i < NN * 64; i += stride) {
    int f = i & 63;
    float v = fmaxf(h[i] * fin[f] + fin[64 + f], 0.0f);
    h[i]  = v;
    hh[i] = (_Float16)v;
  }
}

// -------- per-graph mean pool + final linear (batch is sorted) --------
__global__ __launch_bounds__(256)
void pool_kernel(const float* __restrict__ h, const int* __restrict__ batch,
                 const float* __restrict__ Wro, const float* __restrict__ bro,
                 float* __restrict__ out) {
  const int g = blockIdx.x;
  int lo = 0, hi = NN;
  while (lo < hi) { int mid = (lo + hi) >> 1; if (batch[mid] < g) lo = mid + 1; else hi = mid; }
  const int start = lo;
  lo = 0; hi = NN;
  while (lo < hi) { int mid = (lo + hi) >> 1; if (batch[mid] < g + 1) lo = mid + 1; else hi = mid; }
  const int end = lo;

  const int f  = threadIdx.x & 63;
  const int rl = threadIdx.x >> 6;
  float acc = 0.f;
  for (int n = start + rl; n < end; n += 4) acc += h[n * 64 + f];

  __shared__ float red[256];
  __shared__ float dotp[64];
  red[threadIdx.x] = acc;
  __syncthreads();
  if (threadIdx.x < 64) {
    float sum  = red[f] + red[64 + f] + red[128 + f] + red[192 + f];
    float cnt  = (float)(end - start);
    float mean = sum / fmaxf(cnt, 1.0f);
    dotp[f] = mean * Wro[f];
  }
  __syncthreads();
  if (threadIdx.x == 0) {
    float d = 0.f;
    for (int i = 0; i < 64; ++i) d += dotp[i];
    out[g] = d + bro[0];
  }
}

extern "C" void kernel_launch(void* const* d_in, const int* in_sizes, int n_in,
                              void* d_out, int out_size, void* d_ws, size_t ws_size,
                              hipStream_t stream) {
  const float* x     = (const float*)d_in[0];
  const int*   eidx  = (const int*)d_in[1];
  const float* eattr = (const float*)d_in[2];
  const int*   batch = (const int*)d_in[3];
  const float* Wn1 = (const float*)d_in[4];
  const float* bn1 = (const float*)d_in[5];
  const float* Wr1 = (const float*)d_in[6];
  const float* g1  = (const float*)d_in[7];
  const float* be1 = (const float*)d_in[8];
  const float* Wn2 = (const float*)d_in[9];
  const float* bn2 = (const float*)d_in[10];
  const float* Wr2 = (const float*)d_in[11];
  const float* g2  = (const float*)d_in[12];
  const float* be2 = (const float*)d_in[13];
  const float* Wro = (const float*)d_in[14];
  const float* bro = (const float*)d_in[15];
  float* out = (float*)d_out;
  const int* srcI = eidx;
  const int* dstI = eidx + NE;

  char* ws = (char*)d_ws;
  size_t off = 0;
  auto alloc = [&](size_t bytes) -> void* {
    void* p = ws + off;
    off = (off + bytes + 255) & ~(size_t)255;
    return p;
  };
  float*    sAcc  = (float*)alloc((size_t)NN * 64 * 4);
  float*    hBuf  = (float*)alloc((size_t)NN * 64 * 4);
  _Float16* hh    = (_Float16*)alloc((size_t)NN * 64 * 2);
  float*    deg   = (float*)alloc((size_t)NN * 4);
  float*    stats = (float*)alloc(128 * 4);
  float*    fin   = (float*)alloc(128 * 4);
  _Float16* Wn1T  = (_Float16*)alloc(96 * 64 * 2);
  _Float16* Wn2T  = (_Float16*)alloc(96 * 64 * 2);
  _Float16* Wr1T  = (_Float16*)alloc(64 * 64 * 2);
  _Float16* Wr2T  = (_Float16*)alloc(64 * 64 * 2);

  // ---- setup ----
  prep_weights_kernel<<<24, 256, 0, stream>>>(Wn1, Wn2, Wr1, Wr2, Wn1T, Wn2T, Wr1T, Wr2T);
  cvt_kernel<<<1024, 256, 0, stream>>>(x, hh, NN * 64);
  zero_kernel<<<196, 256, 0, stream>>>(deg, NN);
  deg_kernel<<<(NE + 255) / 256, 256, 0, stream>>>(dstI, deg);

  // ---- layer 1 ----
  zero_kernel<<<2048, 256, 0, stream>>>(sAcc, NN * 64);
  edge_msg_wmma<<<512, 256, 0, stream>>>(hh, eattr, srcI, dstI, Wn1T, bn1, sAcc);
  node_update_wmma<<<256, 256, 0, stream>>>(hh, Wr1T, sAcc, deg, hBuf);
  zero_kernel<<<1, 128, 0, stream>>>(stats, 128);
  bn_stats_kernel<<<512, 256, 0, stream>>>(hBuf, stats);
  bn_final_kernel<<<1, 64, 0, stream>>>(stats, g1, be1, fin);
  bn_apply_kernel<<<1024, 256, 0, stream>>>(hBuf, hh, fin);

  // ---- layer 2 ----
  zero_kernel<<<2048, 256, 0, stream>>>(sAcc, NN * 64);
  edge_msg_wmma<<<512, 256, 0, stream>>>(hh, eattr, srcI, dstI, Wn2T, bn2, sAcc);
  node_update_wmma<<<256, 256, 0, stream>>>(hh, Wr2T, sAcc, deg, hBuf);
  zero_kernel<<<1, 128, 0, stream>>>(stats, 128);
  bn_stats_kernel<<<512, 256, 0, stream>>>(hBuf, stats);
  bn_final_kernel<<<1, 64, 0, stream>>>(stats, g2, be2, fin);
  bn_apply_kernel<<<1024, 256, 0, stream>>>(hBuf, hh, fin);

  // ---- readout ----
  pool_kernel<<<NG, 256, 0, stream>>>(hBuf, batch, Wro, bro, out);
}
